// GVPBackbone_77309411358
// MI455X (gfx1250) — compile-verified, matrix operand order checked by reading, not run on hardware
//
#include <hip/hip_runtime.h>
#include <math.h>

// ---------------- problem constants (match reference) ----------------
#define NN   20000     // nodes
#define EE   320000    // edges
#define BB   32        // graphs
#define SH   128       // node scalar channels
#define VH   16        // node vector channels
#define SE   32        // edge scalar channels
#define OUTC 128

typedef __bf16 v16bf __attribute__((ext_vector_type(16)));
typedef float  v8f   __attribute__((ext_vector_type(8)));
typedef unsigned int v4u __attribute__((ext_vector_type(4)));
typedef int v8i __attribute__((ext_vector_type(8)));
typedef int v4i __attribute__((ext_vector_type(4)));

struct GW {
    const float* wh;
    const float* bs;
    const float* wv;
    const float* wsv;
    const float* bsv;
    const unsigned short* wp;   // bf16 B-fragment packed ws
};

// =====================================================================
// TDM gather: pull 16 rows (width f32 elements each) of a row-major
// [nrows x width] tensor, selected by 16-bit row indices pk[0..7]
// (2 indices per dword), into LDS at lds_addr. Gather-mode D#.
// =====================================================================
static __device__ __forceinline__ void tdm_gather16_rows(
    const float* base, unsigned width, unsigned nrows,
    const unsigned* pk, unsigned lds_addr)
{
    unsigned long long ga = (unsigned long long)(uintptr_t)base;
    // group0: count=1, gather_mode=1 (bit31), 16-bit indices (bit30=0),
    //         lds_addr, global_addr[56:0], type=2 (bits 127:126)
    v4u g0 = { 0x80000001u,
               lds_addr,
               (unsigned)ga,
               ((unsigned)(ga >> 32) & 0x01FFFFFFu) | 0x80000000u };
    // group1: data_size=4B (2<<16); tensor_dim0=width; tensor_dim1=nrows;
    //         tile_dim0=width; tile_dim1=16 indices; dim0 stride=width
    v8i g1 = { (int)(2u << 16),
               (int)(width << 16),
               (int)(nrows << 16),
               (int)(width << 16),
               16,
               (int)width,
               0, 0 };
    v4i g2 = { (int)pk[0], (int)pk[1], (int)pk[2], (int)pk[3] };
    v4i g3 = { (int)pk[4], (int)pk[5], (int)pk[6], (int)pk[7] };
#if defined(__clang_major__) && (__clang_major__ >= 23)
    v8i gz = { 0, 0, 0, 0, 0, 0, 0, 0 };
    __builtin_amdgcn_tensor_load_to_lds(g0, g1, g2, g3, gz, 0);
#else
    __builtin_amdgcn_tensor_load_to_lds(g0, g1, g2, g3, 0);
#endif
}

// =====================================================================
// Generic 16-row GVP tile executed by ONE wave32 (see round-1 comments).
// =====================================================================
static __device__ void gvp_tile(
    int lane, int SI, int VI, int SO, int VO, int H, int KPAD,
    const float* wh, const float* bs, const float* wv,
    const float* wsv, const float* bsv, const unsigned short* wp,
    int relu_scalar,
    __bf16* ain,            // [16][KPAD] bf16 (LDS)
    const float* vin,       // [16][VI][3] f32 (LDS)
    float* vh,              // [16][H][3]  f32 scratch (LDS)
    __bf16* sout_bf, int KOUT,  // optional bf16 scalar output (row stride KOUT)
    float* sout_f,              // optional f32 scalar output [16][SO]
    float* vout,            // [16][VO][3] f32 (LDS)
    float* gbuf)            // [16][VO]    f32 (LDS)
{
    // ---- 1) vh = einsum("ric,hi->rhc", vin, wh) ----
    const int HW = H * 3;
    for (int t = lane; t < 16 * HW; t += 32) {
        int r = t / HW, rem = t % HW, hh = rem / 3, c = rem % 3;
        const float* wrow = wh + hh * VI;
        float acc = 0.f;
        for (int i = 0; i < VI; ++i)
            acc += wrow[i] * vin[(r * VI + i) * 3 + c];
        vh[(r * H + hh) * 3 + c] = acc;
    }
    __syncthreads();

    // ---- 2) vn = ||vh|| into ain cols [SI, KPAD), zero pad ----
    const int tail = KPAD - SI;
    for (int t = lane; t < 16 * tail; t += 32) {
        int r = t / tail, j = t % tail;
        float val = 0.f;
        if (j < H) {
            const float* p = vh + (r * H + j) * 3;
            float ss = p[0]*p[0] + p[1]*p[1] + p[2]*p[2];
            val = sqrtf(fmaxf(ss, 1e-8f));
        }
        ain[r * KPAD + SI + j] = (__bf16)val;
    }
    __syncthreads();

    // ---- 3) s_out = A @ ws^T + bs  via v_wmma_f32_16x16x32_bf16 ----
    const int KT = KPAD >> 5;
    const int NT = SO >> 4;
    const int mrow  = lane & 15;
    const int khalf = (lane >> 4) << 4;
    const int mbase = (lane >> 4) << 3;
    for (int nt = 0; nt < NT; ++nt) {
        v8f acc = {0.f,0.f,0.f,0.f,0.f,0.f,0.f,0.f};
        for (int kt = 0; kt < KT; ++kt) {
            const v16bf a = *(const v16bf*)(ain + mrow * KPAD + kt * 32 + khalf);
            const v16bf b = *(const v16bf*)(wp + (((kt * NT + nt) * 32 + lane) << 4));
            acc = __builtin_amdgcn_wmma_f32_16x16x32_bf16(
                      false, a, false, b, (short)0, acc, false, false);
        }
        const int col  = nt * 16 + (lane & 15);
        const float bias = bs[col];
        if (sout_bf) {
            for (int r2 = 0; r2 < 8; ++r2)
                sout_bf[(mbase + r2) * KOUT + col] = (__bf16)(acc[r2] + bias);
        } else {
            for (int r2 = 0; r2 < 8; ++r2)
                sout_f[(mbase + r2) * SO + col] = acc[r2] + bias;
        }
    }
    __syncthreads();

    // ---- 4) gate = sigmoid(s_out @ wsv^T + bsv) (raw s_out, pre-relu) ----
    for (int t = lane; t < 16 * VO; t += 32) {
        int r = t / VO, o = t % VO;
        const float* wr = wsv + o * SO;
        float acc = bsv[o];
        if (sout_bf) {
            const __bf16* srow = sout_bf + r * KOUT;
            for (int n = 0; n < SO; ++n) acc += wr[n] * (float)srow[n];
        } else {
            const float* srow = sout_f + r * SO;
            for (int n = 0; n < SO; ++n) acc += wr[n] * srow[n];
        }
        gbuf[r * VO + o] = 1.f / (1.f + expf(-acc));
    }
    __syncthreads();

    // ---- 5) v_out = (vh @ wv^T) * gate ----
    const int VW = VO * 3;
    for (int t = lane; t < 16 * VW; t += 32) {
        int r = t / VW, rem = t % VW, o = rem / 3, c = rem % 3;
        const float* wr = wv + o * H;
        float acc = 0.f;
        for (int hh = 0; hh < H; ++hh)
            acc += wr[hh] * vh[(r * H + hh) * 3 + c];
        vout[(r * VO + o) * 3 + c] = acc * gbuf[r * VO + o];
    }

    // ---- 6) relu on scalar output ----
    if (relu_scalar) {
        for (int t = lane; t < 16 * SO; t += 32) {
            int r = t / SO, n = t % SO;
            if (sout_bf) {
                float x = (float)sout_bf[r * KOUT + n];
                sout_bf[r * KOUT + n] = (__bf16)fmaxf(x, 0.f);
            } else {
                sout_f[t] = fmaxf(sout_f[t], 0.f);
            }
        }
    }
    __syncthreads();
}

// ---- tuple LayerNorm on a 16-row tile (lanes 0-15 scalars, 16-31 vectors) --
static __device__ void tile_ln(int lane, float* sbuf, float* vbuf,
                               const float* g, const float* b)
{
    if (lane < 16) {
        const int r = lane;
        float mu = 0.f;
        for (int k = 0; k < SH; ++k) mu += sbuf[r * SH + k];
        mu *= (1.f / SH);
        float var = 0.f;
        for (int k = 0; k < SH; ++k) {
            float d = sbuf[r * SH + k] - mu;
            var += d * d;
        }
        var *= (1.f / SH);
        const float rs = rsqrtf(var + 1e-5f);
        for (int k = 0; k < SH; ++k)
            sbuf[r * SH + k] = (sbuf[r * SH + k] - mu) * rs * g[k] + b[k];
    } else {
        const int r = lane - 16;
        float acc = 0.f;
        for (int ch = 0; ch < VH; ++ch) {
            const float* p = vbuf + (r * VH + ch) * 3;
            acc += fmaxf(p[0]*p[0] + p[1]*p[1] + p[2]*p[2], 1e-8f);
        }
        const float inv = rsqrtf(acc * (1.f / VH));
        for (int j = 0; j < VH * 3; ++j) vbuf[r * VH * 3 + j] *= inv;
    }
}

// =====================================================================
// Edge message kernel: one wave32 per 16-edge tile.
// Node-feature row gathers are done by the Tensor Data Mover (gather-mode
// descriptors, TENSORcnt-tracked); edge features load directly.
// =====================================================================
__global__ __launch_bounds__(32) void edge_msg_kernel(
    const float* __restrict__ s, const float* __restrict__ v,
    const float* __restrict__ es, const float* __restrict__ ev,
    const int* __restrict__ src, const int* __restrict__ dst,
    GW g0, GW g1, GW g2,
    float* __restrict__ ds, float* __restrict__ dv)
{
    __shared__ int sh_src[16], sh_dst[16];
    __shared__ alignas(32) __bf16 shA[16 * 352];
    __shared__ alignas(32) __bf16 shB[16 * 160];
    __shared__ float shV [16 * 33 * 3];
    __shared__ float shVH[16 * 33 * 3];
    __shared__ float shS [16 * 128];
    __shared__ float shG [16 * 16];
    __shared__ alignas(16) float stSs[16 * 128];   // TDM: s[src] rows
    __shared__ alignas(16) float stSd[16 * 128];   // TDM: s[dst] rows
    __shared__ alignas(16) float stVs[16 * 48];    // TDM: v[src] rows
    __shared__ alignas(16) float stVd[16 * 48];    // TDM: v[dst] rows

    const int lane = threadIdx.x;
    const int e0   = blockIdx.x << 4;

    // pack 16-bit row indices (uniform addresses -> SGPRs) for the D#
    unsigned pkS[8], pkD[8];
    for (int j = 0; j < 8; ++j) {
        pkS[j] = ((unsigned)src[e0 + 2*j] & 0xFFFFu) |
                 (((unsigned)src[e0 + 2*j + 1] & 0xFFFFu) << 16);
        pkD[j] = ((unsigned)dst[e0 + 2*j] & 0xFFFFu) |
                 (((unsigned)dst[e0 + 2*j + 1] & 0xFFFFu) << 16);
    }
    // issue the four DMA gathers
    tdm_gather16_rows(s, SH,     NN, pkS, (unsigned)(uintptr_t)stSs);
    tdm_gather16_rows(s, SH,     NN, pkD, (unsigned)(uintptr_t)stSd);
    tdm_gather16_rows(v, VH * 3, NN, pkS, (unsigned)(uintptr_t)stVs);
    tdm_gather16_rows(v, VH * 3, NN, pkD, (unsigned)(uintptr_t)stVd);

    if (lane < 16) { sh_src[lane] = src[e0 + lane]; sh_dst[lane] = dst[e0 + lane]; }

    // edge scalars -> bf16 while DMA is in flight
    for (int t = lane; t < 16 * 32; t += 32) {
        int r = t >> 5, k = t & 31;
        shA[r * 352 + 128 + k] = (__bf16)es[(e0 + r) * SE + k];
    }

    __builtin_amdgcn_s_wait_tensorcnt(0);
    __syncthreads();

    // staged f32 rows -> bf16 A matrix: [s_src(128) | edge_s(32) | s_dst(128)]
    for (int t = lane; t < 16 * 128; t += 32) {
        int r = t >> 7, k = t & 127;
        shA[r * 352 + k]       = (__bf16)stSs[t];
        shA[r * 352 + 160 + k] = (__bf16)stSd[t];
    }
    // vectors: [v_src(16) | edge_v(1) | v_dst(16)]
    for (int t = lane; t < 16 * 99; t += 32) {
        int r = t / 99, rem = t % 99, ch = rem / 3, c = rem % 3;
        float val;
        if (ch < 16)       val = stVs[r * 48 + ch * 3 + c];
        else if (ch == 16) val = ev[(e0 + r) * 3 + c];
        else               val = stVd[r * 48 + (ch - 17) * 3 + c];
        shV[(r * 33 + ch) * 3 + c] = val;
    }
    __syncthreads();

    // GVP0: (288,33) -> (128,16), relu
    gvp_tile(lane, 288, 33, 128, 16, 33, 352,
             g0.wh, g0.bs, g0.wv, g0.wsv, g0.bsv, g0.wp, 1,
             shA, shV, shVH, shB, 160, nullptr, shV, shG);
    // GVP1: (128,16) -> (128,16), relu
    gvp_tile(lane, 128, 16, 128, 16, 16, 160,
             g1.wh, g1.bs, g1.wv, g1.wsv, g1.bsv, g1.wp, 1,
             shB, shV, shVH, shA, 160, nullptr, shV, shG);
    // GVP2: (128,16) -> (128,16), no relu, f32 out
    gvp_tile(lane, 128, 16, 128, 16, 16, 160,
             g2.wh, g2.bs, g2.wv, g2.wsv, g2.bsv, g2.wp, 0,
             shA, shV, shVH, nullptr, 0, shS, shV, shG);

    // scatter-add messages to destination nodes
    for (int t = lane; t < 16 * 128; t += 32) {
        int r = t >> 7;
        atomicAdd(ds + (long)sh_dst[r] * SH + (t & 127), shS[t]);
    }
    for (int t = lane; t < 16 * 48; t += 32) {
        int r = t / 48;
        atomicAdd(dv + (long)sh_dst[r] * (VH * 3) + (t % 48), shV[t]);
    }
}

// =====================================================================
// Node update kernel: residual + LN0, FF (2 GVPs), residual + LN1.
// =====================================================================
__global__ __launch_bounds__(32) void node_ff_kernel(
    float* __restrict__ s, float* __restrict__ v,
    const float* __restrict__ ds, const float* __restrict__ dv,
    const float* __restrict__ cnt,
    GW f0, GW f1,
    const float* __restrict__ ln0g, const float* __restrict__ ln0b,
    const float* __restrict__ ln1g, const float* __restrict__ ln1b)
{
    __shared__ float shSres[16 * 128];
    __shared__ float shVres[16 * 48];
    __shared__ alignas(32) __bf16 shA[16 * 544];
    __shared__ alignas(32) __bf16 shB[16 * 160];
    __shared__ float shV [16 * 32 * 3];
    __shared__ float shVH[16 * 32 * 3];
    __shared__ float shS [16 * 128];
    __shared__ float shG [16 * 32];

    const int lane = threadIdx.x;
    const int n0   = blockIdx.x << 4;

    for (int t = lane; t < 16 * 128; t += 32) {
        int r = t >> 7;
        float c = fmaxf(cnt[n0 + r], 1.f);
        shSres[t] = s[n0 * SH + t] + ds[n0 * SH + t] / c;
    }
    for (int t = lane; t < 16 * 48; t += 32) {
        int r = t / 48;
        float c = fmaxf(cnt[n0 + r], 1.f);
        shVres[t] = v[n0 * (VH * 3) + t] + dv[n0 * (VH * 3) + t] / c;
    }
    __syncthreads();
    tile_ln(lane, shSres, shVres, ln0g, ln0b);
    __syncthreads();

    for (int t = lane; t < 16 * 128; t += 32)
        shB[(t >> 7) * 160 + (t & 127)] = (__bf16)shSres[t];
    __syncthreads();

    // FF0: (128,16) -> (512,32), relu
    gvp_tile(lane, 128, 16, 512, 32, 32, 160,
             f0.wh, f0.bs, f0.wv, f0.wsv, f0.bsv, f0.wp, 1,
             shB, shVres, shVH, shA, 544, nullptr, shV, shG);
    // FF1: (512,32) -> (128,16), no relu, f32 out
    gvp_tile(lane, 512, 32, 128, 16, 32, 544,
             f1.wh, f1.bs, f1.wv, f1.wsv, f1.bsv, f1.wp, 0,
             shA, shV, shVH, nullptr, 0, shS, shV, shG);

    for (int t = lane; t < 16 * 128; t += 32) shSres[t] += shS[t];
    for (int t = lane; t < 16 * 48;  t += 32) shVres[t] += shV[t];
    __syncthreads();
    tile_ln(lane, shSres, shVres, ln1g, ln1b);
    __syncthreads();

    for (int t = lane; t < 16 * 128; t += 32) s[n0 * SH + t]       = shSres[t];
    for (int t = lane; t < 16 * 48;  t += 32) v[n0 * (VH * 3) + t] = shVres[t];
}

// =====================================================================
// Utility kernels
// =====================================================================
__global__ void pack_ws_kernel(const float* __restrict__ ws, int K, int SO,
                               unsigned short* __restrict__ out, int total)
{
    int t = blockIdx.x * blockDim.x + threadIdx.x;
    if (t >= total) return;
    const int NT = SO >> 4;
    int inner = t & 511;
    int blk   = t >> 9;
    int nt = blk % NT;
    int kt = blk / NT;
    int lane = inner >> 4;
    int i    = inner & 15;
    int k = kt * 32 + ((lane >> 4) << 4) + i;
    int n = nt * 16 + (lane & 15);
    float v = (k < K) ? ws[n * K + k] : 0.f;
    __bf16 h = (__bf16)v;
    out[t] = __builtin_bit_cast(unsigned short, h);
}

__global__ void init_kernel(const float* __restrict__ xs, const float* __restrict__ xv,
                            float* s, float* v, float* ds, float* dv,
                            float* cnt, float* gcnt, float* gpool)
{
    int i = blockIdx.x * blockDim.x + threadIdx.x;
    if (i < NN * SH)      { s[i] = xs[i]; ds[i] = 0.f; }
    if (i < NN * VH * 3)  { v[i] = xv[i]; dv[i] = 0.f; }
    if (i < NN)           cnt[i] = 0.f;
    if (i < BB * SH)      gpool[i] = 0.f;
    if (i < BB)           gcnt[i] = 0.f;
}

__global__ void zero_dsdv_kernel(float* ds, float* dv)
{
    int i = blockIdx.x * blockDim.x + threadIdx.x;
    if (i < NN * SH)     ds[i] = 0.f;
    if (i < NN * VH * 3) dv[i] = 0.f;
}

__global__ void count_kernel(const int* __restrict__ dst, float* cnt)
{
    int i = blockIdx.x * blockDim.x + threadIdx.x;
    if (i < EE) atomicAdd(cnt + dst[i], 1.f);
}

__global__ void gcount_kernel(const int* __restrict__ batch, float* gcnt)
{
    int i = blockIdx.x * blockDim.x + threadIdx.x;
    if (i < NN) atomicAdd(gcnt + batch[i], 1.f);
}

__global__ void pool_kernel(const float* __restrict__ s,
                            const int* __restrict__ batch, float* gpool)
{
    int i = blockIdx.x * blockDim.x + threadIdx.x;
    if (i < NN * SH)
        atomicAdd(gpool + batch[i >> 7] * SH + (i & 127), s[i]);
}

__global__ __launch_bounds__(128) void head_kernel(
    const float* __restrict__ gpool, const float* __restrict__ gcnt,
    const float* __restrict__ w1, const float* __restrict__ b1,
    const float* __restrict__ w2, const float* __restrict__ b2,
    float* __restrict__ out)
{
    __shared__ float gbuf[SH];
    __shared__ float hbuf[SH];
    const int b = blockIdx.x;
    const int j = threadIdx.x;
    gbuf[j] = gpool[b * SH + j] / fmaxf(gcnt[b], 1.f);
    __syncthreads();
    float a = b1[j];
    for (int i = 0; i < SH; ++i) a += gbuf[i] * w1[i * SH + j];
    hbuf[j] = fmaxf(a, 0.f);
    __syncthreads();
    float o = b2[j];
    for (int i = 0; i < SH; ++i) o += hbuf[i] * w2[i * OUTC + j];
    out[b * OUTC + j] = o;
}

// =====================================================================
// Host launcher
// =====================================================================
extern "C" void kernel_launch(void* const* d_in, const int* in_sizes, int n_in,
                              void* d_out, int out_size, void* d_ws, size_t ws_size,
                              hipStream_t stream)
{
    (void)in_sizes; (void)n_in; (void)out_size; (void)ws_size;

    char* w = (char*)d_ws;
    auto carve = [&](size_t bytes) -> void* {
        void* p = (void*)w;
        w += (bytes + 255) & ~(size_t)255;
        return p;
    };

    float* scur  = (float*)carve((size_t)NN * SH * 4);
    float* vcur  = (float*)carve((size_t)NN * VH * 3 * 4);
    float* ds    = (float*)carve((size_t)NN * SH * 4);
    float* dv    = (float*)carve((size_t)NN * VH * 3 * 4);
    float* cnt   = (float*)carve((size_t)NN * 4);
    float* gcnt  = (float*)carve((size_t)BB * 4);
    float* gpool = (float*)carve((size_t)BB * SH * 4);

    static const int KP[5]   = {352, 160, 160, 160, 544};
    static const int KR[5]   = {321, 144, 144, 160, 544};
    static const int SOsz[5] = {128, 128, 128, 512, 128};
    unsigned short* wpk[3][5];
    for (int l = 0; l < 3; ++l)
        for (int j = 0; j < 5; ++j)
            wpk[l][j] = (unsigned short*)carve((size_t)KP[j] * SOsz[j] * 2);

    auto F = [&](int i) { return (const float*)d_in[i]; };
    const float* xs  = F(0);
    const float* xv  = F(1);
    const float* es  = F(2);
    const float* ev  = F(3);
    const int* srcP  = (const int*)d_in[4];
    const int* dstP  = srcP + EE;
    const int* batch = (const int*)d_in[5];

    for (int l = 0; l < 3; ++l) {
        const int base = 6 + 34 * l;
        const int wsIdx[5] = {base + 1, base + 7, base + 13, base + 19, base + 25};
        for (int j = 0; j < 5; ++j) {
            int total = KP[j] * SOsz[j];
            pack_ws_kernel<<<(total + 255) / 256, 256, 0, stream>>>(
                F(wsIdx[j]), KR[j], SOsz[j], wpk[l][j], total);
        }
    }

    init_kernel<<<(NN * SH + 255) / 256, 256, 0, stream>>>(
        xs, xv, scur, vcur, ds, dv, cnt, gcnt, gpool);
    count_kernel<<<(EE + 255) / 256, 256, 0, stream>>>(dstP, cnt);
    gcount_kernel<<<(NN + 255) / 256, 256, 0, stream>>>(batch, gcnt);

    for (int l = 0; l < 3; ++l) {
        const int base = 6 + 34 * l;
        if (l > 0)
            zero_dsdv_kernel<<<(NN * SH + 255) / 256, 256, 0, stream>>>(ds, dv);

        GW m0{F(base + 0),  F(base + 2),  F(base + 3),  F(base + 4),  F(base + 5),  wpk[l][0]};
        GW m1{F(base + 6),  F(base + 8),  F(base + 9),  F(base + 10), F(base + 11), wpk[l][1]};
        GW m2{F(base + 12), F(base + 14), F(base + 15), F(base + 16), F(base + 17), wpk[l][2]};
        GW f0{F(base + 18), F(base + 20), F(base + 21), F(base + 22), F(base + 23), wpk[l][3]};
        GW f1{F(base + 24), F(base + 26), F(base + 27), F(base + 28), F(base + 29), wpk[l][4]};

        edge_msg_kernel<<<EE / 16, 32, 0, stream>>>(
            scur, vcur, es, ev, srcP, dstP, m0, m1, m2, ds, dv);

        node_ff_kernel<<<NN / 16, 32, 0, stream>>>(
            scur, vcur, ds, dv, cnt, f0, f1,
            F(base + 30), F(base + 31), F(base + 32), F(base + 33));
    }

    pool_kernel<<<(NN * SH + 255) / 256, 256, 0, stream>>>(scur, batch, gpool);
    head_kernel<<<BB, 128, 0, stream>>>(
        gpool, gcnt, F(108), F(109), F(110), F(111), (float*)d_out);
}
